// EncodeBlock_33397665693949
// MI455X (gfx1250) — matrix operand order repaired and checked by reading, not compiled, not run on hardware
//
#include <hip/hip_runtime.h>
#include <stdint.h>

// ---------------------------------------------------------------------------
// EncodeBlock for MI455X (gfx1250): bf16 WMMA, fp32 accumulate.
// No softmax => (QK^T)V == Q(K^T V): per-(b,h) 64x64 inner matrix.
// All GEMM B-operands pre-transposed to [N][K] so LDS staging is pure b128.
// Double-buffered LDS pipeline; async global->LDS DMA when toolchain has it.
// ---------------------------------------------------------------------------

typedef __bf16 bf16x16 __attribute__((ext_vector_type(16)));
typedef float  v8f     __attribute__((ext_vector_type(8)));
union Frag { uint4 u4[2]; bf16x16 v; };

#if __has_builtin(__builtin_amdgcn_global_load_async_to_lds_b128)
#define HAVE_ASYNC_LDS 1
#else
#define HAVE_ASYNC_LDS 0
#endif

// exact parameter types per hipcc diagnostic: vector_size(16) int, AS1 / AS3
typedef int v4i_vs __attribute__((vector_size(16)));
typedef __attribute__((address_space(1))) v4i_vs* gv4p;
typedef __attribute__((address_space(3))) v4i_vs* lv4p;

__device__ __forceinline__ void async_cp16(const unsigned short* g, unsigned short* l) {
#if HAVE_ASYNC_LDS
    // flat LDS pointer low 32 bits == LDS byte offset (ISA 10.2 aperture rule),
    // so the integer-cast is equivalent to the generic->AS(3) addrspacecast.
    __builtin_amdgcn_global_load_async_to_lds_b128(
        (gv4p)(uintptr_t)g, (lv4p)(uintptr_t)l, 0, 0);
#else
    *(uint4*)l = *(const uint4*)g;
#endif
}

__device__ __forceinline__ void wait_async0() {
#if HAVE_ASYNC_LDS
#if __has_builtin(__builtin_amdgcn_s_wait_asynccnt)
    __builtin_amdgcn_s_wait_asynccnt(0);
#else
    asm volatile("s_wait_asynccnt 0x0" ::: "memory");
#endif
#endif
}

__device__ __forceinline__ unsigned short f2bf(float f) {
    union { float f; unsigned u; } x; x.f = f;
    unsigned r = x.u + 0x7FFFu + ((x.u >> 16) & 1u);   // round-to-nearest-even
    return (unsigned short)(r >> 16);
}

#define WMMA_BF16(A, B, C) \
    __builtin_amdgcn_wmma_f32_16x16x32_bf16(false, (A), false, (B), (short)0, (C), false, false)

#define GPAD 40   // bf16 elems per LDS row (80B) -> conflict-free b128 lane reads

// ---------------------------------------------------------------------------
// converters (one-time, tiny vs GEMM cost)
// ---------------------------------------------------------------------------
__global__ __launch_bounds__(256) void cvt_plain_kernel(const float* __restrict__ in,
                                                        unsigned short* __restrict__ out, int n) {
    int i = blockIdx.x * 256 + threadIdx.x;
    if (i < n) out[i] = f2bf(in[i]);
}

// weight [K][N] fp32 -> transposed [N][K] bf16
__global__ __launch_bounds__(256) void cvt_wT_kernel(const float* __restrict__ in,
                                                     unsigned short* __restrict__ out, int K, int N) {
    int i = blockIdx.x * 256 + threadIdx.x;
    if (i >= N * K) return;
    int n = i / K, k = i - n * K;
    out[i] = f2bf(in[(size_t)k * N + n]);
}

// wq/wk/wv [H, D, HD] fp32 -> transposed concat [N=H*HD][K=D] bf16
__global__ __launch_bounds__(256) void cvt_headwT_kernel(const float* __restrict__ in,
                                                         unsigned short* __restrict__ out) {
    int i = blockIdx.x * 256 + threadIdx.x;       // i = n*1024 + k
    int n = i >> 10, k = i & 1023;
    int h = n >> 6, e = n & 63;
    out[i] = f2bf(in[((h << 10) + k) * 64 + e]);
}

// ---------------------------------------------------------------------------
// Tiled bf16 GEMM: C[M,N] = A[M,K] @ Bt[N,K]^T (+bias)(relu).
// 128x128 block, BK=32, 8 waves (wave tile 64x32). Double-buffered LDS with
// async global->LDS DMA (or register-staged pipeline fallback).
// Cb optionally stored transposed [N][M] as packed b128 (for K/V).
// ---------------------------------------------------------------------------
__global__ __launch_bounds__(256)
void gemm_bf16_kernel(const unsigned short* __restrict__ A, int lda,
                      const unsigned short* __restrict__ Bt, int ldb,
                      const float* __restrict__ bias,
                      float* __restrict__ Cf, unsigned short* __restrict__ Cb,
                      int ldc, int ldcb, int cbT,
                      int M, int N, int K, int relu)
{
    __shared__ unsigned short As[2][128 * GPAD];
    __shared__ unsigned short Bs[2][128 * GPAD];
    (void)M;

    const int tid  = threadIdx.x;
    const int wave = tid >> 5, lane = tid & 31;
    const int lm = lane & 15, khalf = lane >> 4;
    const int m0 = blockIdx.y * 128, n0 = blockIdx.x * 128;
    const int wm = (wave >> 2) * 64, wn = (wave & 3) * 32;

    // staging: 512 b128 chunks per 128x32 tile, 2 per thread for A and for B
    const int cr = tid >> 2, ck = (tid & 3) * 8;
    const unsigned short* gA0 = A  + (size_t)(m0 + cr)      * lda + ck;
    const unsigned short* gA1 = A  + (size_t)(m0 + cr + 64) * lda + ck;
    const unsigned short* gB0 = Bt + (size_t)(n0 + cr)      * ldb + ck;
    const unsigned short* gB1 = Bt + (size_t)(n0 + cr + 64) * ldb + ck;
    const int l0 = cr * GPAD + ck, l1 = (cr + 64) * GPAD + ck;

#if HAVE_ASYNC_LDS
    auto issue = [&](int k0, int buf) {
        async_cp16(gA0 + k0, &As[buf][l0]);
        async_cp16(gA1 + k0, &As[buf][l1]);
        async_cp16(gB0 + k0, &Bs[buf][l0]);
        async_cp16(gB1 + k0, &Bs[buf][l1]);
    };
    issue(0, 0);
#else
    uint4 ra0, ra1, rb0, rb1;
    auto fetch = [&](int k0) {
        ra0 = *(const uint4*)(gA0 + k0);
        ra1 = *(const uint4*)(gA1 + k0);
        rb0 = *(const uint4*)(gB0 + k0);
        rb1 = *(const uint4*)(gB1 + k0);
    };
    auto commit = [&](int buf) {
        *(uint4*)&As[buf][l0] = ra0;
        *(uint4*)&As[buf][l1] = ra1;
        *(uint4*)&Bs[buf][l0] = rb0;
        *(uint4*)&Bs[buf][l1] = rb1;
    };
    fetch(0);
#endif

    v8f acc[4][2] = {};
    const int nsteps = K >> 5;
    for (int s = 0; s < nsteps; ++s) {
        const int buf = s & 1;
#if HAVE_ASYNC_LDS
        wait_async0();
        __syncthreads();
        if (s + 1 < nsteps) issue((s + 1) << 5, buf ^ 1);
#else
        commit(buf);
        __syncthreads();
        if (s + 1 < nsteps) fetch((s + 1) << 5);     // overlaps with WMMAs below
#endif
        Frag bf2[2];
        #pragma unroll
        for (int j = 0; j < 2; ++j) {
            const uint4* p = (const uint4*)&Bs[buf][(wn + j * 16 + lm) * GPAD];
            bf2[j].u4[0] = p[khalf]; bf2[j].u4[1] = p[2 + khalf];
        }
        #pragma unroll
        for (int i = 0; i < 4; ++i) {
            Frag af;
            const int mrow = (wm + i * 16 + lm) * GPAD;
            af.u4[0] = *(const uint4*)&As[buf][mrow + khalf * 8];
            af.u4[1] = *(const uint4*)&As[buf][mrow + 16 + khalf * 8];
            #pragma unroll
            for (int j = 0; j < 2; ++j)
                acc[i][j] = WMMA_BF16(af.v, bf2[j].v, acc[i][j]);
        }
    }

    // epilogue: C vgpr r -> (row = khalf*8 + r, col = lm) within each 16x16 tile
    #pragma unroll
    for (int i = 0; i < 4; ++i) {
        #pragma unroll
        for (int j = 0; j < 2; ++j) {
            const int gcol  = n0 + wn + j * 16 + lm;
            const int grow0 = m0 + wm + i * 16 + khalf * 8;
            const float bv = bias ? bias[gcol] : 0.f;
            float v[8];
            #pragma unroll
            for (int r = 0; r < 8; ++r) {
                float t = acc[i][j][r] + bv;
                v[r] = relu ? fmaxf(t, 0.f) : t;
            }
            if (Cf) {
                #pragma unroll
                for (int r = 0; r < 8; ++r)
                    Cf[(size_t)(grow0 + r) * ldc + gcol] = v[r];
            }
            if (Cb) {
                if (cbT) {   // transposed [N][M]: 8 contiguous rows -> one b128
                    unsigned pk0 = (unsigned)f2bf(v[0]) | ((unsigned)f2bf(v[1]) << 16);
                    unsigned pk1 = (unsigned)f2bf(v[2]) | ((unsigned)f2bf(v[3]) << 16);
                    unsigned pk2 = (unsigned)f2bf(v[4]) | ((unsigned)f2bf(v[5]) << 16);
                    unsigned pk3 = (unsigned)f2bf(v[6]) | ((unsigned)f2bf(v[7]) << 16);
                    uint4 q; q.x = pk0; q.y = pk1; q.z = pk2; q.w = pk3;
                    *(uint4*)(Cb + (size_t)gcol * ldcb + grow0) = q;
                } else {
                    #pragma unroll
                    for (int r = 0; r < 8; ++r)
                        Cb[(size_t)(grow0 + r) * ldc + gcol] = f2bf(v[r]);
                }
            }
        }
    }
}

// ---------------------------------------------------------------------------
// Mt = scale * (K^T V) stored transposed (Mt[e*64+f]), per (b,h).
// Inputs are K^T/V^T bf16 [1024][4096] so staging is pure b128.
// ---------------------------------------------------------------------------
__global__ __launch_bounds__(256)
void ktv_kernel(const unsigned short* __restrict__ KbT, const unsigned short* __restrict__ VbT,
                unsigned short* __restrict__ Mt, float scale)
{
    __shared__ unsigned short Kt[2][64 * GPAD];   // [f][t]
    __shared__ unsigned short Vt[2][64 * GPAD];   // [e][t]

    const int bh = blockIdx.x, b = bh >> 4, h = bh & 15;
    const int tid = threadIdx.x, wave = tid >> 5, lane = tid & 31;
    const int lm = lane & 15, khalf = lane >> 4;
    const int ftile = wave & 3, eg = wave >> 2;   // wave: 1 f-tile x 2 e-tiles

    const int cf = tid >> 2, ck = (tid & 3) * 8;  // 256 chunks per 64x32 tile
    const unsigned short* gK = KbT + (size_t)(h * 64 + cf) * 4096 + b * 2048 + ck;
    const unsigned short* gV = VbT + (size_t)(h * 64 + cf) * 4096 + b * 2048 + ck;
    const int loff = cf * GPAD + ck;

#if HAVE_ASYNC_LDS
    auto issue = [&](int t0, int buf) {
        async_cp16(gK + t0, &Kt[buf][loff]);
        async_cp16(gV + t0, &Vt[buf][loff]);
    };
    issue(0, 0);
#else
    uint4 rk, rv;
    auto fetch  = [&](int t0)  { rk = *(const uint4*)(gK + t0); rv = *(const uint4*)(gV + t0); };
    auto commit = [&](int buf) { *(uint4*)&Kt[buf][loff] = rk; *(uint4*)&Vt[buf][loff] = rv; };
    fetch(0);
#endif

    v8f acc[2] = {};
    for (int s = 0; s < 64; ++s) {
        const int buf = s & 1;
#if HAVE_ASYNC_LDS
        wait_async0();
        __syncthreads();
        if (s + 1 < 64) issue((s + 1) << 5, buf ^ 1);
#else
        commit(buf);
        __syncthreads();
        if (s + 1 < 64) fetch((s + 1) << 5);
#endif
        Frag af;
        const int fr = (ftile * 16 + lm) * GPAD;
        af.u4[0] = *(const uint4*)&Kt[buf][fr + khalf * 8];
        af.u4[1] = *(const uint4*)&Kt[buf][fr + 16 + khalf * 8];
        #pragma unroll
        for (int j = 0; j < 2; ++j) {
            Frag bv;
            const uint4* p = (const uint4*)&Vt[buf][((2 * eg + j) * 16 + lm) * GPAD];
            bv.u4[0] = p[khalf]; bv.u4[1] = p[2 + khalf];
            acc[j] = WMMA_BF16(af.v, bv.v, acc[j]);
        }
    }

    unsigned short* out = Mt + (size_t)bh * 4096;
    #pragma unroll
    for (int j = 0; j < 2; ++j) {
        const int e = (2 * eg + j) * 16 + lm;
        #pragma unroll
        for (int r = 0; r < 8; ++r)
            out[e * 64 + ftile * 16 + khalf * 8 + r] = f2bf(acc[j][r] * scale);  // transposed
    }
}

// ---------------------------------------------------------------------------
// attn = Q_h @ M_h -> cat (bf16). K=64 in registers, operands straight from
// global (M_h is 8KB, L0-resident). grid = (T/128, B*H), wave = 16x64 tile.
// ---------------------------------------------------------------------------
__global__ __launch_bounds__(256)
void attn_kernel(const unsigned short* __restrict__ Qb, const unsigned short* __restrict__ Mt,
                 unsigned short* __restrict__ catb)
{
    const int bh = blockIdx.y, b = bh >> 4, h = bh & 15;
    const int tid = threadIdx.x, wave = tid >> 5, lane = tid & 31;
    const int lm = lane & 15, khalf = lane >> 4;
    const int trow = blockIdx.x * 128 + wave * 16;

    const unsigned short* Qg = Qb + ((size_t)(b * 2048 + trow)) * 1024 + h * 64;
    const unsigned short* Mg = Mt + (size_t)bh * 4096;

    Frag bf[2][4];
    #pragma unroll
    for (int ks = 0; ks < 2; ++ks)
        #pragma unroll
        for (int j = 0; j < 4; ++j) {
            const uint4* p = (const uint4*)(Mg + (j * 16 + lm) * 64 + ks * 32);
            bf[ks][j].u4[0] = p[khalf]; bf[ks][j].u4[1] = p[2 + khalf];
        }

    v8f acc[4] = {};
    #pragma unroll
    for (int ks = 0; ks < 2; ++ks) {
        Frag af;
        af.u4[0] = *(const uint4*)(Qg + (size_t)lm * 1024 + ks * 32 + khalf * 8);
        af.u4[1] = *(const uint4*)(Qg + (size_t)lm * 1024 + ks * 32 + 16 + khalf * 8);
        #pragma unroll
        for (int j = 0; j < 4; ++j)
            acc[j] = WMMA_BF16(af.v, bf[ks][j].v, acc[j]);
    }

    unsigned short* out = catb + ((size_t)(b * 2048 + trow)) * 1024 + h * 64;
    #pragma unroll
    for (int j = 0; j < 4; ++j)
        #pragma unroll
        for (int r = 0; r < 8; ++r)
            out[(size_t)(khalf * 8 + r) * 1024 + j * 16 + lm] = f2bf(acc[j][r]);
}

// ---------------------------------------------------------------------------
// out = LayerNorm(a + b)*g + beta; row staged in LDS (aliasing-safe).
// ---------------------------------------------------------------------------
__global__ __launch_bounds__(256)
void ln_kernel(const float* __restrict__ a, const float* __restrict__ b,
               const float* __restrict__ g, const float* __restrict__ beta,
               float* __restrict__ outf, unsigned short* __restrict__ outb)
{
    __shared__ float row[1024];
    __shared__ float red[256];
    const int r = blockIdx.x, tid = threadIdx.x;
    const float* ap = a + (size_t)r * 1024;
    const float* bp = b ? b + (size_t)r * 1024 : nullptr;

    float s = 0.f;
    #pragma unroll
    for (int i = 0; i < 4; ++i) {
        int c = tid + 256 * i;
        float v = ap[c] + (bp ? bp[c] : 0.f);
        row[c] = v; s += v;
    }
    red[tid] = s; __syncthreads();
    for (int st = 128; st > 0; st >>= 1) { if (tid < st) red[tid] += red[tid + st]; __syncthreads(); }
    const float mu = red[0] * (1.f / 1024.f);
    __syncthreads();

    float vs = 0.f;
    #pragma unroll
    for (int i = 0; i < 4; ++i) { float d = row[tid + 256 * i] - mu; vs += d * d; }
    red[tid] = vs; __syncthreads();
    for (int st = 128; st > 0; st >>= 1) { if (tid < st) red[tid] += red[tid + st]; __syncthreads(); }
    const float rstd = rsqrtf(red[0] * (1.f / 1024.f) + 1e-5f);
    __syncthreads();

    #pragma unroll
    for (int i = 0; i < 4; ++i) {
        int c = tid + 256 * i;
        float o = (row[c] - mu) * rstd * g[c] + beta[c];
        outf[(size_t)r * 1024 + c] = o;
        if (outb) outb[(size_t)r * 1024 + c] = f2bf(o);
    }
}

// ---------------------------------------------------------------------------
// Host launcher
// ---------------------------------------------------------------------------
extern "C" void kernel_launch(void* const* d_in, const int* in_sizes, int n_in,
                              void* d_out, int out_size, void* d_ws, size_t ws_size,
                              hipStream_t stream)
{
    (void)in_sizes; (void)n_in; (void)out_size; (void)ws_size;
    const int NT = 4096, Dm = 1024, D4 = 4096;
    const size_t NTD = (size_t)NT * Dm;

    const float* x    = (const float*)d_in[0];
    const float* wq   = (const float*)d_in[1];
    const float* bq   = (const float*)d_in[2];
    const float* wk   = (const float*)d_in[3];
    const float* bk   = (const float*)d_in[4];
    const float* wv   = (const float*)d_in[5];
    const float* bv   = (const float*)d_in[6];
    const float* wp   = (const float*)d_in[7];
    const float* bp   = (const float*)d_in[8];
    const float* w1   = (const float*)d_in[9];
    const float* b1   = (const float*)d_in[10];
    const float* w2   = (const float*)d_in[11];
    const float* b2   = (const float*)d_in[12];
    const float* ln1g = (const float*)d_in[13];
    const float* ln1b = (const float*)d_in[14];
    const float* lnfg = (const float*)d_in[15];
    const float* lnfb = (const float*)d_in[16];
    const float* ln2g = (const float*)d_in[17];
    const float* ln2b = (const float*)d_in[18];

    float* out2 = (float*)d_out;
    float* Kc   = out2 + NTD;
    float* Vc   = out2 + 2 * NTD;

    char* ws = (char*)d_ws;
    size_t off = 0;
    auto carve = [&](size_t bytes) { char* p = ws + off; off += (bytes + 255) & ~(size_t)255; return p; };
    unsigned short* xb    = (unsigned short*)carve(NTD * 2);
    unsigned short* Wq    = (unsigned short*)carve((size_t)Dm * Dm * 2);   // [N][K]
    unsigned short* Wk    = (unsigned short*)carve((size_t)Dm * Dm * 2);
    unsigned short* Wv    = (unsigned short*)carve((size_t)Dm * Dm * 2);
    unsigned short* Wp    = (unsigned short*)carve((size_t)Dm * Dm * 2);
    unsigned short* W1    = (unsigned short*)carve((size_t)Dm * D4 * 2);   // [4096][1024]
    unsigned short* W2    = (unsigned short*)carve((size_t)D4 * Dm * 2);   // [1024][4096]
    unsigned short* Qb    = (unsigned short*)carve(NTD * 2);               // row-major [NT][D]
    unsigned short* KbT   = (unsigned short*)carve(NTD * 2);               // transposed [D][NT]
    unsigned short* VbT   = (unsigned short*)carve(NTD * 2);               // transposed [D][NT]
    unsigned short* Mt    = (unsigned short*)carve((size_t)32 * 64 * 64 * 2);
    unsigned short* catb  = (unsigned short*)carve(NTD * 2);
    float*          out1f = (float*)carve(NTD * 4);
    unsigned short* out1b = (unsigned short*)carve(NTD * 2);
    unsigned short* h1b   = (unsigned short*)carve((size_t)NT * D4 * 2);
    float*          prjff = (float*)carve(NTD * 4);    // proj, reused as ff/ffo

    // 1) convert (weights pre-transposed to [N][K])
    cvt_plain_kernel<<<(int)(NTD / 256), 256, 0, stream>>>(x, xb, (int)NTD);
    cvt_headwT_kernel<<<Dm * Dm / 256, 256, 0, stream>>>(wq, Wq);
    cvt_headwT_kernel<<<Dm * Dm / 256, 256, 0, stream>>>(wk, Wk);
    cvt_headwT_kernel<<<Dm * Dm / 256, 256, 0, stream>>>(wv, Wv);
    cvt_wT_kernel<<<Dm * Dm / 256, 256, 0, stream>>>(wp, Wp, Dm, Dm);
    cvt_wT_kernel<<<Dm * D4 / 256, 256, 0, stream>>>(w1, W1, Dm, D4);
    cvt_wT_kernel<<<D4 * Dm / 256, 256, 0, stream>>>(w2, W2, D4, Dm);

    // 2) QKV projections; K/V f32 to d_out, bf16 copies transposed for K^T V
    dim3 g1(Dm / 128, NT / 128);
    gemm_bf16_kernel<<<g1, 256, 0, stream>>>(xb, Dm, Wq, Dm, bq, nullptr, Qb,  Dm, 0,  0, NT, Dm, Dm, 0);
    gemm_bf16_kernel<<<g1, 256, 0, stream>>>(xb, Dm, Wk, Dm, bk, Kc,      KbT, Dm, NT, 1, NT, Dm, Dm, 0);
    gemm_bf16_kernel<<<g1, 256, 0, stream>>>(xb, Dm, Wv, Dm, bv, Vc,      VbT, Dm, NT, 1, NT, Dm, Dm, 0);

    // 3) linear re-association: Mt = scale*(K^T V) (transposed), attn = Q @ M
    ktv_kernel<<<32, 256, 0, stream>>>(KbT, VbT, Mt, 0.125f);            // 64^-0.5
    attn_kernel<<<dim3(16, 32), 256, 0, stream>>>(Qb, Mt, catb);

    // 4) output projection + residual layernorm
    gemm_bf16_kernel<<<g1, 256, 0, stream>>>(catb, Dm, Wp, Dm, bp, prjff, nullptr, Dm, 0, 0, NT, Dm, Dm, 0);
    ln_kernel<<<NT, 256, 0, stream>>>(x, prjff, ln1g, ln1b, out1f, out1b);

    // 5) FFN
    dim3 g2(D4 / 128, NT / 128);
    gemm_bf16_kernel<<<g2, 256, 0, stream>>>(out1b, Dm, W1, Dm, b1, nullptr, h1b, D4, 0, 0, NT, D4, Dm, 1);
    gemm_bf16_kernel<<<g1, 256, 0, stream>>>(h1b,   D4, W2, D4, b2, prjff, nullptr, Dm, 0, 0, NT, Dm, D4, 0);

    // 6) ffo = LN(out1 + ff) (aliasing-safe), out2 = LN(out1 + ffo)
    ln_kernel<<<NT, 256, 0, stream>>>(out1f, prjff, lnfg, lnfb, prjff, nullptr);
    ln_kernel<<<NT, 256, 0, stream>>>(out1f, prjff, ln2g, ln2b, out2,  nullptr);
}
